// LSTMCell_704374636891
// MI455X (gfx1250) — matrix operand order repaired and checked by reading, not compiled, not run on hardware
//
#include <hip/hip_runtime.h>

// LSTM cell, fused 4-gate GEMM + elementwise, bf16 WMMA, gfx1250 (wave32).
// B=4096, DX=DH=4096, DO=4096, K = DX+DH = 8192.
//
// Two-phase path (when ws_size allows):
//   Phase 1: convert inputs -> bf16 [m][k]; convert+transpose weights -> bf16 [g][n][k].
//   Phase 2: LDS-free GEMM: WMMA fragments loaded directly from global bf16 (L2-resident reuse).
// Fallback path (small ws): single-phase LDS-staged kernel with inline f32->bf16 conversion.

#define BM 128            // block tile M (both paths)
#define BN 64             // block tile N per gate (both paths)
#define BK 32             // k-step (= one bf16 wmma K)
#define NG 4              // gates: f, g, i, o
#define LDA 40            // fallback LDS row stride (bf16 elems)
#define LDB 40
#define DXC 4096
#define NPG 4096          // per-gate output columns
#define KTOT 8192
#define MTOT 4096
#define KTILES (KTOT / BK)

typedef __attribute__((ext_vector_type(16))) __bf16 v16bf;
typedef __attribute__((ext_vector_type(8)))  float  v8f;

union FragU { uint2 u2[4]; uint4 u4[2]; v16bf v; };
union F4    { float4 v; float f[4]; };

__device__ __forceinline__ unsigned short f2bf(float f) {
  unsigned u = __float_as_uint(f);
  unsigned r = u + 0x7FFFu + ((u >> 16) & 1u);   // round-to-nearest-even
  return (unsigned short)(r >> 16);
}
__device__ __forceinline__ unsigned bfpack(float a, float b) {
  return (unsigned)f2bf(a) | ((unsigned)f2bf(b) << 16);
}
__device__ __forceinline__ float fsigmoid(float x) {
  return 1.0f / (1.0f + __expf(-x));
}
__device__ __forceinline__ float ftanh(float x) {
  float t = __expf(-2.0f * fabsf(x));            // stable tanh
  float r = (1.0f - t) / (1.0f + t);
  return copysignf(r, x);
}

// ---------------------------------------------------------------------------
// Phase 1a: concat(x,h) f32 -> bf16 [m][KTOT], k-contiguous. 8 elems/thread.
// ---------------------------------------------------------------------------
__global__ __launch_bounds__(256) void cvt_inputs(
    const float* __restrict__ x, const float* __restrict__ h,
    unsigned short* __restrict__ abf)
{
  const unsigned idx = blockIdx.x * 256 + threadIdx.x;   // one per 8 elems
  const unsigned k8  = idx & 1023;                       // KTOT/8 = 1024
  const unsigned m   = idx >> 10;
  const unsigned k   = k8 * 8;                           // 8 | DXC: no straddle
  const float* src = (k < DXC) ? (x + (size_t)m * DXC + k)
                               : (h + (size_t)m * DXC + (k - DXC));
  F4 v0, v1;
  v0.v = *(const float4*)(src);
  v1.v = *(const float4*)(src + 4);
  uint4 pk;
  pk.x = bfpack(v0.f[0], v0.f[1]);
  pk.y = bfpack(v0.f[2], v0.f[3]);
  pk.z = bfpack(v1.f[0], v1.f[1]);
  pk.w = bfpack(v1.f[2], v1.f[3]);
  *(uint4*)(abf + (size_t)m * KTOT + k) = pk;
}

// ---------------------------------------------------------------------------
// Phase 1b: W[g] f32 [k][n] -> bf16 [g][n][k] (transposed), 32x32 tiles in LDS.
// grid: (KTOT/32, NPG/32, NG), 256 threads.
// ---------------------------------------------------------------------------
__global__ __launch_bounds__(256) void cvt_weights(
    const float* __restrict__ Wf, const float* __restrict__ Wg,
    const float* __restrict__ Wi, const float* __restrict__ Wo,
    unsigned short* __restrict__ bbf)
{
  __shared__ unsigned short t[32 * 40];                  // [k][n], padded
  const float* W[NG] = {Wf, Wg, Wi, Wo};
  const int g  = blockIdx.z;
  const int kt = blockIdx.x * 32;
  const int nt = blockIdx.y * 32;
  const int tid = threadIdx.x;

  // load 32x32 f32 (k rows, n cols), convert, store to LDS [k][n]
  {
    const int r  = tid >> 3;        // k row 0..31
    const int c4 = tid & 7;         // n col quad
    F4 v; v.v = *(const float4*)(W[g] + (size_t)(kt + r) * NPG + nt + c4 * 4);
    uint2 pk; pk.x = bfpack(v.f[0], v.f[1]); pk.y = bfpack(v.f[2], v.f[3]);
    *(uint2*)(&t[r * 40 + c4 * 4]) = pk;
  }
  __syncthreads();
  // read transposed: each thread emits 4 bf16 along k at fixed n
  {
    const int n  = tid & 31;
    const int kc = tid >> 5;        // 0..7 -> k = kc*4
    uint2 pk;
    pk.x = (unsigned)t[(kc * 4 + 0) * 40 + n] | ((unsigned)t[(kc * 4 + 1) * 40 + n] << 16);
    pk.y = (unsigned)t[(kc * 4 + 2) * 40 + n] | ((unsigned)t[(kc * 4 + 3) * 40 + n] << 16);
    *(uint2*)(bbf + ((size_t)g * NPG + nt + n) * KTOT + kt + kc * 4) = pk;
  }
}

// ---------------------------------------------------------------------------
// Phase 2: LDS-free fused GEMM + LSTM epilogue.
// 128 threads = 4 waves; wave tile 64M x 32N x 4 gates (32 wmma / k-step).
// Fragments loaded directly from bf16 global (fragment-native k-contiguous).
// ---------------------------------------------------------------------------
__global__ __launch_bounds__(128) void lstm_gemm_bf16(
    const unsigned short* __restrict__ abf, const unsigned short* __restrict__ bbf,
    const float* __restrict__ c,
    const float* __restrict__ bf, const float* __restrict__ bg,
    const float* __restrict__ bi, const float* __restrict__ bo,
    float* __restrict__ out)
{
  const int tid   = threadIdx.x;
  const int mBase = blockIdx.x * BM;   // m fastest-varying -> weight-slice reuse in L2
  const int nBase = blockIdx.y * BN;

  const int lane = tid & 31;
  const int wave = tid >> 5;
  const int wm   = wave >> 1;          // 0..1 -> M offset wm*64
  const int wn   = wave & 1;           // 0..1 -> N offset wn*32
  const int lr   = lane & 15;
  const int lh   = lane >> 4;

  // Per-lane base pointers (bf16), fragment-native layouts:
  // A frag: v0-3 = 16B @ (m, kh), v4-7 = 16B @ (m, kh+16); kh = lh*8
  const unsigned short* aPtr[4];
  #pragma unroll
  for (int tm = 0; tm < 4; ++tm)
    aPtr[tm] = abf + (size_t)(mBase + wm * 64 + tm * 16 + lr) * KTOT + lh * 8;
  // B frag: v0-3 = 16B @ (n, kh), v4-7 = 16B @ (n, kh+16); kh = lh*8
  const unsigned short* bPtr[NG][2];
  #pragma unroll
  for (int g = 0; g < NG; ++g)
    #pragma unroll
    for (int tn = 0; tn < 2; ++tn)
      bPtr[g][tn] = bbf + ((size_t)g * NPG + nBase + wn * 32 + tn * 16 + lr) * KTOT + lh * 8;

  v8f acc[NG][4][2];
  #pragma unroll
  for (int g = 0; g < NG; ++g)
    #pragma unroll
    for (int tm = 0; tm < 4; ++tm)
      #pragma unroll
      for (int tn = 0; tn < 2; ++tn)
        acc[g][tm][tn] = 0;

  for (int it = 0; it < KTILES; ++it) {
    const int off = it * BK;                  // bf16 elements
    FragU a[4];
    #pragma unroll
    for (int tm = 0; tm < 4; ++tm) {
      a[tm].u4[0] = *(const uint4*)(aPtr[tm] + off);
      a[tm].u4[1] = *(const uint4*)(aPtr[tm] + off + 16);
    }
    #pragma unroll
    for (int g = 0; g < NG; ++g) {
      #pragma unroll
      for (int tn = 0; tn < 2; ++tn) {
        FragU b;
        b.u4[0] = *(const uint4*)(bPtr[g][tn] + off);
        b.u4[1] = *(const uint4*)(bPtr[g][tn] + off + 16);
        #pragma unroll
        for (int tm = 0; tm < 4; ++tm) {
          acc[g][tm][tn] = __builtin_amdgcn_wmma_f32_16x16x32_bf16(
              false, a[tm].v, false, b.v, (short)0, acc[g][tm][tn], false, false);
        }
      }
    }
  }

  // Epilogue: bias + activations + cell update.
  #pragma unroll
  for (int tm = 0; tm < 4; ++tm) {
    #pragma unroll
    for (int tn = 0; tn < 2; ++tn) {
      const int col = nBase + wn * 32 + tn * 16 + lr;
      const float bfv = bf[col], bgv = bg[col], biv = bi[col], bov = bo[col];
      const int rbase = mBase + wm * 64 + tm * 16 + lh * 8;
      #pragma unroll
      for (int v = 0; v < 8; ++v) {
        const int row = rbase + v;
        const float ft = fsigmoid(acc[0][tm][tn][v] + bfv);
        const float gt = ftanh   (acc[1][tm][tn][v] + bgv);
        const float itg= fsigmoid(acc[2][tm][tn][v] + biv);
        const float ot = fsigmoid(acc[3][tm][tn][v] + bov);
        const size_t idx = (size_t)row * NPG + col;
        const float cn = c[idx] * ft + gt * itg;
        out[idx] = cn;                                         // c_out
        out[(size_t)MTOT * NPG + idx] = ftanh(cn) * ot;        // h_out
      }
    }
  }
}

// ---------------------------------------------------------------------------
// Fallback: single-phase LDS-staged kernel (inline conversion), 8 waves/block.
// ---------------------------------------------------------------------------
__global__ __launch_bounds__(256) void lstm_fused_wmma(
    const float* __restrict__ x,  const float* __restrict__ h,  const float* __restrict__ c,
    const float* __restrict__ Wf, const float* __restrict__ bf,
    const float* __restrict__ Wg, const float* __restrict__ bg,
    const float* __restrict__ Wi, const float* __restrict__ bi,
    const float* __restrict__ Wo, const float* __restrict__ bo,
    float* __restrict__ out)
{
  __shared__ unsigned short As[2][BM * LDA];
  __shared__ unsigned short Bs[2][NG * BN * LDB];

  const int tid   = threadIdx.x;
  const int mBase = blockIdx.x * BM;
  const int nBase = blockIdx.y * BN;

  const float* W[NG] = {Wf, Wg, Wi, Wo};

  const int ac4 = tid & 7;
  const int ar0 = tid >> 3;
  const int bid = tid & 127;
  const int gp  = tid >> 7;
  const int bkb = bid >> 4;
  const int bnb = bid & 15;

  auto stage = [&](int it, int buf) {
    const int kt = it * BK;
    const float* asrc = (kt < DXC) ? (x + kt) : (h + (kt - DXC));
    #pragma unroll
    for (int p = 0; p < 4; ++p) {
      int r = ar0 + p * 32;
      F4 v; v.v = *(const float4*)(asrc + (size_t)(mBase + r) * DXC + ac4 * 4);
      uint2 pk; pk.x = bfpack(v.f[0], v.f[1]); pk.y = bfpack(v.f[2], v.f[3]);
      *(uint2*)(&As[buf][r * LDA + ac4 * 4]) = pk;
    }
    #pragma unroll
    for (int gg = 0; gg < 2; ++gg) {
      const int g = gp * 2 + gg;
      const float* wsrc = W[g];
      F4 rw[4];
      #pragma unroll
      for (int j = 0; j < 4; ++j)
        rw[j].v = *(const float4*)(wsrc + (size_t)(kt + bkb * 4 + j) * NPG + nBase + bnb * 4);
      #pragma unroll
      for (int i = 0; i < 4; ++i) {
        uint2 pk;
        pk.x = bfpack(rw[0].f[i], rw[1].f[i]);
        pk.y = bfpack(rw[2].f[i], rw[3].f[i]);
        *(uint2*)(&Bs[buf][(g * BN + bnb * 4 + i) * LDB + bkb * 4]) = pk;
      }
    }
  };

  const int lane = tid & 31;
  const int wave = tid >> 5;
  const int wm   = wave >> 1;
  const int wn   = wave & 1;
  const int lr   = lane & 15;
  const int lh   = lane >> 4;

  v8f acc[NG][2][2];
  #pragma unroll
  for (int g = 0; g < NG; ++g)
    #pragma unroll
    for (int tm = 0; tm < 2; ++tm)
      #pragma unroll
      for (int tn = 0; tn < 2; ++tn)
        acc[g][tm][tn] = 0;

  stage(0, 0);
  __syncthreads();

  for (int it = 0; it < KTILES; ++it) {
    const int buf = it & 1;
    if (it + 1 < KTILES) stage(it + 1, buf ^ 1);

    FragU a[2];
    #pragma unroll
    for (int tm = 0; tm < 2; ++tm) {
      const unsigned short* ab = &As[buf][(wm * 32 + tm * 16 + lr) * LDA];
      const int kh = lh * 8;
      a[tm].u2[0] = *(const uint2*)(ab + kh);
      a[tm].u2[1] = *(const uint2*)(ab + kh + 4);
      a[tm].u2[2] = *(const uint2*)(ab + kh + 16);
      a[tm].u2[3] = *(const uint2*)(ab + kh + 20);
    }
    #pragma unroll
    for (int g = 0; g < NG; ++g) {
      #pragma unroll
      for (int tn = 0; tn < 2; ++tn) {
        const unsigned short* bb =
            &Bs[buf][(g * BN + wn * 32 + tn * 16 + lr) * LDB + lh * 8];
        FragU b;
        b.u2[0] = *(const uint2*)(bb + 0);
        b.u2[1] = *(const uint2*)(bb + 4);
        b.u2[2] = *(const uint2*)(bb + 16);
        b.u2[3] = *(const uint2*)(bb + 20);
        #pragma unroll
        for (int tm = 0; tm < 2; ++tm) {
          acc[g][tm][tn] = __builtin_amdgcn_wmma_f32_16x16x32_bf16(
              false, a[tm].v, false, b.v, (short)0, acc[g][tm][tn], false, false);
        }
      }
    }
    __syncthreads();
  }

  #pragma unroll
  for (int tm = 0; tm < 2; ++tm) {
    #pragma unroll
    for (int tn = 0; tn < 2; ++tn) {
      const int col = nBase + wn * 32 + tn * 16 + lr;
      const float bfv = bf[col], bgv = bg[col], biv = bi[col], bov = bo[col];
      const int rbase = mBase + wm * 32 + tm * 16 + lh * 8;
      #pragma unroll
      for (int v = 0; v < 8; ++v) {
        const int row = rbase + v;
        const float ft = fsigmoid(acc[0][tm][tn][v] + bfv);
        const float gt = ftanh   (acc[1][tm][tn][v] + bgv);
        const float itg= fsigmoid(acc[2][tm][tn][v] + biv);
        const float ot = fsigmoid(acc[3][tm][tn][v] + bov);
        const size_t idx = (size_t)row * NPG + col;
        const float cn = c[idx] * ft + gt * itg;
        out[idx] = cn;
        out[(size_t)MTOT * NPG + idx] = ftanh(cn) * ot;
      }
    }
  }
}

extern "C" void kernel_launch(void* const* d_in, const int* in_sizes, int n_in,
                              void* d_out, int out_size, void* d_ws, size_t ws_size,
                              hipStream_t stream) {
  const float* x  = (const float*)d_in[0];
  const float* h  = (const float*)d_in[1];
  const float* c  = (const float*)d_in[2];
  const float* Wf = (const float*)d_in[3];
  const float* bf = (const float*)d_in[4];
  const float* Wg = (const float*)d_in[5];
  const float* bg = (const float*)d_in[6];
  const float* Wi = (const float*)d_in[7];
  const float* bi = (const float*)d_in[8];
  const float* Wo = (const float*)d_in[9];
  const float* bo = (const float*)d_in[10];
  float* out = (float*)d_out;

  const size_t bytesA = (size_t)MTOT * KTOT * sizeof(unsigned short);        // 64 MB
  const size_t bytesB = (size_t)NG * NPG * KTOT * sizeof(unsigned short);    // 256 MB

  if (ws_size >= bytesA + bytesB) {
    unsigned short* abf = (unsigned short*)d_ws;
    unsigned short* bbf = (unsigned short*)((char*)d_ws + bytesA);

    // Phase 1: one-time conversions (each element touched exactly once).
    cvt_inputs<<<dim3((MTOT * KTOT / 8) / 256), dim3(256), 0, stream>>>(x, h, abf);
    cvt_weights<<<dim3(KTOT / 32, NPG / 32, NG), dim3(256), 0, stream>>>(Wf, Wg, Wi, Wo, bbf);

    // Phase 2: LDS-free WMMA GEMM + fused LSTM epilogue.
    lstm_gemm_bf16<<<dim3(MTOT / BM, NPG / BN), dim3(128), 0, stream>>>(
        abf, bbf, c, bf, bg, bi, bo, out);
  } else {
    // Fallback: single-phase LDS-staged kernel.
    lstm_fused_wmma<<<dim3(MTOT / BM, NPG / BN), dim3(256), 0, stream>>>(
        x, h, c, Wf, bf, Wg, bg, Wi, bi, Wo, bo, out);
  }
}
